// EGNNLayer_67413806678103
// MI455X (gfx1250) — matrix operand order repaired and checked by reading, not compiled, not run on hardware
//
#include <hip/hip_runtime.h>
#include <hip/hip_bf16.h>
#include <math.h>

typedef __attribute__((ext_vector_type(16))) __bf16 v16bf;
typedef __attribute__((ext_vector_type(8)))  __bf16 v8bf;
typedef __attribute__((ext_vector_type(8)))  float  v8f;

#define HD 128
#define PI_F 3.14159265358979323846f

__device__ __forceinline__ float siluf(float x) { return x / (1.f + __expf(-x)); }

// ---- quantum circuit helpers: all statevector indices compile-time ----
template<int B>
__device__ __forceinline__ void qry(float* st, float c, float s) {
#pragma unroll
  for (int i = 0; i < 64; ++i) {
    if (((i >> B) & 1) == 0) {
      const int j = i | (1 << B);
      float a0 = st[i], a1 = st[j];
      st[i] = c * a0 - s * a1;
      st[j] = s * a0 + c * a1;
    }
  }
}
template<int CB, int TB>
__device__ __forceinline__ void qcnot(float* st) {
#pragma unroll
  for (int i = 0; i < 64; ++i) {
    if (((i >> CB) & 1) && (((i >> TB) & 1) == 0)) {
      const int j = i | (1 << TB);
      float t = st[i]; st[i] = st[j]; st[j] = t;
    }
  }
}
__device__ __forceinline__ void ry_layer(float* st, const float* a) {
  float c, s;
  c = __cosf(0.5f * a[0]); s = __sinf(0.5f * a[0]); qry<5>(st, c, s); // qubit 0 = bit 5
  c = __cosf(0.5f * a[1]); s = __sinf(0.5f * a[1]); qry<4>(st, c, s);
  c = __cosf(0.5f * a[2]); s = __sinf(0.5f * a[2]); qry<3>(st, c, s);
  c = __cosf(0.5f * a[3]); s = __sinf(0.5f * a[3]); qry<2>(st, c, s);
  c = __cosf(0.5f * a[4]); s = __sinf(0.5f * a[4]); qry<1>(st, c, s);
  c = __cosf(0.5f * a[5]); s = __sinf(0.5f * a[5]); qry<0>(st, c, s);
}

// Assemble 16x32 bf16 A-fragment from a row of LDS (contiguous k order: element s holds k = ab + s).
__device__ __forceinline__ v16bf mk_afrag(const __bf16* rowp, int ab) {
  v8bf lo = *(const v8bf*)(rowp + ab);        // K = ab .. ab+7
  v8bf hi = *(const v8bf*)(rowp + ab + 8);    // K = ab+8 .. ab+15
  v16bf a;
#pragma unroll
  for (int i = 0; i < 8; ++i) { a[i] = lo[i]; a[i + 8] = hi[i]; }
  return a;
}

// ---- weight prep: fp32 [K][128] -> bf16 transposed [col][Kpad] ----
__global__ void prep_kernel(const float* __restrict__ mW1, const float* __restrict__ mW2,
                            const float* __restrict__ nW1, const float* __restrict__ nW2,
                            __bf16* __restrict__ mW1t, __bf16* __restrict__ mW2t,
                            __bf16* __restrict__ nW1t, __bf16* __restrict__ nW2t) {
  int i = blockIdx.x * blockDim.x + threadIdx.x;
  if (i < 36864) {                        // mW1t: [128][288], rows 257..287 zero
    int col = i / 288, k = i % 288;
    mW1t[i] = (__bf16)((k < 257) ? mW1[k * HD + col] : 0.f);
  } else if (i < 53248) {                 // mW2t: [128][128]
    int j = i - 36864; int col = j / 128, k = j % 128;
    mW2t[j] = (__bf16)mW2[k * HD + col];
  } else if (i < 86016) {                 // nW1t: [128][256]
    int j = i - 53248; int col = j / 256, k = j % 256;
    nW1t[j] = (__bf16)nW1[k * HD + col];
  } else if (i < 102400) {                // nW2t: [128][128]
    int j = i - 86016; int col = j / 128, k = j % 128;
    nW2t[j] = (__bf16)nW2[k * HD + col];
  }
}

__global__ void zero_kernel(float* p, long n) {
  long i = (long)blockIdx.x * blockDim.x + threadIdx.x;
  long stride = (long)gridDim.x * blockDim.x;
  for (; i < n; i += stride) p[i] = 0.f;
}

// ---- edge pipeline: gather -> WMMA MLP -> quantum -> atomic aggregation ----
__global__ void __launch_bounds__(32)
egnn_edge_kernel(const float* __restrict__ h, const float* __restrict__ pos,
                 const int* __restrict__ ei,
                 const float* __restrict__ mb1, const float* __restrict__ mb2,
                 const float* __restrict__ pqW, const float* __restrict__ pqb,
                 const float* __restrict__ qw,
                 const float* __restrict__ postW, const float* __restrict__ postb,
                 const __bf16* __restrict__ mW1t, const __bf16* __restrict__ mW2t,
                 float* __restrict__ agg_msg, float* __restrict__ agg_trans, int E) {
  // Abuf (16x288 bf16, GEMM1 input) and Mbuf (16x132 f32, post-GEMM2 msg tile)
  // are live in disjoint phases -> overlay to cut LDS per workgroup.
  __shared__ __align__(32) unsigned char smem0[16 * 288 * 2];
  __bf16* Abuf = (__bf16*)smem0;
  float*  Mbuf = (float*)smem0;           // 16*132*4 = 8448 B <= 9216 B
  __shared__ __align__(32) __bf16 Hbuf[16 * HD];
  __shared__ int    sRow[16], sCol[16], sValid[16];
  __shared__ float  sCD[16 * 3], sRad[16];

  const int lane = threadIdx.x;
  const int halfSel = lane >> 4;
  const int mn = lane & 15;
  const int e0 = blockIdx.x * 16;

  if (lane < 16) {
    int e = e0 + lane;
    int valid = (e < E) ? 1 : 0;
    int ec = valid ? e : (E - 1);
    int r = ei[ec], c = ei[E + ec];
    sRow[lane] = r; sCol[lane] = c; sValid[lane] = valid;
    float dx = pos[r * 3 + 0] - pos[c * 3 + 0];
    float dy = pos[r * 3 + 1] - pos[c * 3 + 1];
    float dz = pos[r * 3 + 2] - pos[c * 3 + 2];
    sCD[lane * 3 + 0] = dx; sCD[lane * 3 + 1] = dy; sCD[lane * 3 + 2] = dz;
    sRad[lane] = dx * dx + dy * dy + dz * dz + 1e-8f;
  }
  __syncthreads();

  for (int m = 0; m < 16; ++m) {
    const int r = sRow[m], c = sCol[m], valid = sValid[m];
    const float rad = sRad[m];
#pragma unroll
    for (int t = 0; t < 9; ++t) {
      int k = lane + 32 * t;
      float v = 0.f;
      if (valid) {
        if (k < HD)           v = h[r * HD + k];
        else if (k < 2 * HD)  v = h[c * HD + (k - HD)];
        else if (k == 2 * HD) v = rad;
      }
      Abuf[m * 288 + k] = (__bf16)v;
    }
  }
  __syncthreads();

  // GEMM1: [16,288] x [288,128] (K padded, radial folded in).
  // Preload all 8 B fragments per K-chunk so loads clause together and the
  // 8 WMMAs issue back-to-back behind a single wait.
  v8f acc[8];
#pragma unroll
  for (int nt = 0; nt < 8; ++nt)
#pragma unroll
    for (int j = 0; j < 8; ++j) acc[nt][j] = 0.f;

#pragma unroll
  for (int kc = 0; kc < 9; ++kc) {
    const int ab = kc * 32 + (halfSel ? 16 : 0);
    const int kb = kc * 32 + (halfSel ? 16 : 0);
    v16bf bF[8];
#pragma unroll
    for (int nt = 0; nt < 8; ++nt)
      bF[nt] = *(const v16bf*)&mW1t[(nt * 16 + mn) * 288 + kb];
    v16bf aF = mk_afrag(&Abuf[mn * 288], ab);
#pragma unroll
    for (int nt = 0; nt < 8; ++nt)
      acc[nt] = __builtin_amdgcn_wmma_f32_16x16x32_bf16(false, aF, false, bF[nt],
                                                        (short)0, acc[nt], false, false);
  }

#pragma unroll
  for (int nt = 0; nt < 8; ++nt) {
    const int col = nt * 16 + mn;
    const float b1 = mb1[col];
#pragma unroll
    for (int v = 0; v < 8; ++v) {
      const int row = v + halfSel * 8;
      Hbuf[row * HD + col] = (__bf16)siluf(acc[nt][v] + b1);
    }
  }
  __syncthreads();

  // GEMM2: [16,128] x [128,128]
  v8f acc2[8];
#pragma unroll
  for (int nt = 0; nt < 8; ++nt)
#pragma unroll
    for (int j = 0; j < 8; ++j) acc2[nt][j] = 0.f;

#pragma unroll
  for (int kc = 0; kc < 4; ++kc) {
    const int ab = kc * 32 + (halfSel ? 16 : 0);
    const int kb = kc * 32 + (halfSel ? 16 : 0);
    v16bf bF[8];
#pragma unroll
    for (int nt = 0; nt < 8; ++nt)
      bF[nt] = *(const v16bf*)&mW2t[(nt * 16 + mn) * HD + kb];
    v16bf aF = mk_afrag(&Hbuf[mn * HD], ab);
#pragma unroll
    for (int nt = 0; nt < 8; ++nt)
      acc2[nt] = __builtin_amdgcn_wmma_f32_16x16x32_bf16(false, aF, false, bF[nt],
                                                         (short)0, acc2[nt], false, false);
  }

#pragma unroll
  for (int nt = 0; nt < 8; ++nt) {
    const int col = nt * 16 + mn;
    const float b2 = mb2[col];
#pragma unroll
    for (int v = 0; v < 8; ++v) {
      const int row = v + halfSel * 8;
      const float msg = acc2[nt][v] + b2;
      Mbuf[row * 132 + col] = msg;
      if (sValid[row]) atomicAdd(&agg_msg[sRow[row] * HD + col], msg);
    }
  }
  __syncthreads();

  // quantum edge update: one edge per lane, statevector in registers
  if (lane < 16 && sValid[lane]) {
    float ang[6];
#pragma unroll
    for (int q = 0; q < 6; ++q) {
      float s = pqb[q];
#pragma unroll 8
      for (int k = 0; k < HD; ++k) s = fmaf(Mbuf[lane * 132 + k], pqW[k * 6 + q], s);
      ang[q] = tanhf(s) * PI_F;
    }
    float st[64];
#pragma unroll
    for (int i = 0; i < 64; ++i) st[i] = 0.f;
    st[0] = 1.f;
    ry_layer(st, ang);
    float w[6];
#pragma unroll
    for (int l = 0; l < 2; ++l) {
#pragma unroll
      for (int q = 0; q < 6; ++q) w[q] = qw[l * 6 + q];
      ry_layer(st, w);
      qcnot<5, 4>(st); qcnot<4, 3>(st); qcnot<3, 2>(st);
      qcnot<2, 1>(st); qcnot<1, 0>(st); qcnot<0, 5>(st);
    }
    float p = 0.f;
#pragma unroll
    for (int i = 0; i < 32; ++i) p += st[i] * st[i];
#pragma unroll
    for (int i = 32; i < 64; ++i) p -= st[i] * st[i];
    const float cw = p * postW[0] + postb[0];
    const int r = sRow[lane];
    atomicAdd(&agg_trans[r * 3 + 0], sCD[lane * 3 + 0] * cw);
    atomicAdd(&agg_trans[r * 3 + 1], sCD[lane * 3 + 1] * cw);
    atomicAdd(&agg_trans[r * 3 + 2], sCD[lane * 3 + 2] * cw);
  }
}

// ---- node update: [h | agg_msg] -> MLP (WMMA) -> residual ----
__global__ void __launch_bounds__(32)
egnn_node_kernel(const float* __restrict__ h, const float* __restrict__ agg_msg,
                 const float* __restrict__ nb1, const float* __restrict__ nb2,
                 const __bf16* __restrict__ nW1t, const __bf16* __restrict__ nW2t,
                 float* __restrict__ h_out, int N) {
  __shared__ __align__(32) __bf16 Abuf[16 * 256];
  __shared__ __align__(32) __bf16 Hbuf[16 * HD];
  const int lane = threadIdx.x;
  const int halfSel = lane >> 4;
  const int mn = lane & 15;
  const int n0 = blockIdx.x * 16;

  for (int m = 0; m < 16; ++m) {
    int node = n0 + m;
    int valid = node < N;
    int idx = valid ? node : (N - 1);
#pragma unroll
    for (int t = 0; t < 8; ++t) {
      int k = lane + 32 * t;
      float v = 0.f;
      if (valid) v = (k < HD) ? h[idx * HD + k] : agg_msg[idx * HD + (k - HD)];
      Abuf[m * 256 + k] = (__bf16)v;
    }
  }
  __syncthreads();

  v8f acc[8];
#pragma unroll
  for (int nt = 0; nt < 8; ++nt)
#pragma unroll
    for (int j = 0; j < 8; ++j) acc[nt][j] = 0.f;

#pragma unroll
  for (int kc = 0; kc < 8; ++kc) {
    const int ab = kc * 32 + (halfSel ? 16 : 0);
    const int kb = kc * 32 + (halfSel ? 16 : 0);
    v16bf bF[8];
#pragma unroll
    for (int nt = 0; nt < 8; ++nt)
      bF[nt] = *(const v16bf*)&nW1t[(nt * 16 + mn) * 256 + kb];
    v16bf aF = mk_afrag(&Abuf[mn * 256], ab);
#pragma unroll
    for (int nt = 0; nt < 8; ++nt)
      acc[nt] = __builtin_amdgcn_wmma_f32_16x16x32_bf16(false, aF, false, bF[nt],
                                                        (short)0, acc[nt], false, false);
  }

#pragma unroll
  for (int nt = 0; nt < 8; ++nt) {
    const int col = nt * 16 + mn;
    const float b1 = nb1[col];
#pragma unroll
    for (int v = 0; v < 8; ++v) {
      const int row = v + halfSel * 8;
      Hbuf[row * HD + col] = (__bf16)siluf(acc[nt][v] + b1);
    }
  }
  __syncthreads();

  v8f acc2[8];
#pragma unroll
  for (int nt = 0; nt < 8; ++nt)
#pragma unroll
    for (int j = 0; j < 8; ++j) acc2[nt][j] = 0.f;

#pragma unroll
  for (int kc = 0; kc < 4; ++kc) {
    const int ab = kc * 32 + (halfSel ? 16 : 0);
    const int kb = kc * 32 + (halfSel ? 16 : 0);
    v16bf bF[8];
#pragma unroll
    for (int nt = 0; nt < 8; ++nt)
      bF[nt] = *(const v16bf*)&nW2t[(nt * 16 + mn) * HD + kb];
    v16bf aF = mk_afrag(&Hbuf[mn * HD], ab);
#pragma unroll
    for (int nt = 0; nt < 8; ++nt)
      acc2[nt] = __builtin_amdgcn_wmma_f32_16x16x32_bf16(false, aF, false, bF[nt],
                                                         (short)0, acc2[nt], false, false);
  }

#pragma unroll
  for (int nt = 0; nt < 8; ++nt) {
    const int col = nt * 16 + mn;
    const float b2 = nb2[col];
#pragma unroll
    for (int v = 0; v < 8; ++v) {
      const int row = v + halfSel * 8;
      const int node = n0 + row;
      if (node < N) h_out[node * HD + col] = h[node * HD + col] + acc2[nt][v] + b2;
    }
  }
}

__global__ void pos_kernel(const float* __restrict__ pos, const float* __restrict__ agg_trans,
                           float* __restrict__ out, int n3) {
  int i = blockIdx.x * blockDim.x + threadIdx.x;
  if (i < n3) out[i] = pos[i] + agg_trans[i];
}

extern "C" void kernel_launch(void* const* d_in, const int* in_sizes, int n_in,
                              void* d_out, int out_size, void* d_ws, size_t ws_size,
                              hipStream_t stream) {
  const float* h     = (const float*)d_in[0];
  const float* pos   = (const float*)d_in[1];
  const int*   ei    = (const int*)d_in[2];
  const float* mW1   = (const float*)d_in[3];
  const float* mb1   = (const float*)d_in[4];
  const float* mW2   = (const float*)d_in[5];
  const float* mb2   = (const float*)d_in[6];
  const float* pqW   = (const float*)d_in[7];
  const float* pqb   = (const float*)d_in[8];
  const float* qw    = (const float*)d_in[9];
  const float* postW = (const float*)d_in[10];
  const float* postb = (const float*)d_in[11];
  const float* nW1   = (const float*)d_in[12];
  const float* nb1   = (const float*)d_in[13];
  const float* nW2   = (const float*)d_in[14];
  const float* nb2   = (const float*)d_in[15];

  const int N = in_sizes[0] / HD;
  const int E = in_sizes[2] / 2;

  char* ws = (char*)d_ws;
  __bf16* mW1t = (__bf16*)(ws);                   // 128*288*2 = 73728 B
  __bf16* mW2t = (__bf16*)(ws + 73728);           // 32768 B
  __bf16* nW1t = (__bf16*)(ws + 106496);          // 65536 B
  __bf16* nW2t = (__bf16*)(ws + 172032);          // 32768 B
  float*  agg_msg   = (float*)(ws + 204800);      // N*128 floats
  float*  agg_trans = agg_msg + (size_t)N * HD;   // N*3 floats, contiguous

  prep_kernel<<<(102400 + 255) / 256, 256, 0, stream>>>(mW1, mW2, nW1, nW2,
                                                        mW1t, mW2t, nW1t, nW2t);
  long zn = (long)N * HD + (long)N * 3;
  zero_kernel<<<2048, 256, 0, stream>>>(agg_msg, zn);

  int etiles = (E + 15) / 16;
  egnn_edge_kernel<<<etiles, 32, 0, stream>>>(h, pos, ei, mb1, mb2, pqW, pqb, qw,
                                              postW, postb, mW1t, mW2t,
                                              agg_msg, agg_trans, E);
  int ntiles = (N + 15) / 16;
  egnn_node_kernel<<<ntiles, 32, 0, stream>>>(h, agg_msg, nb1, nb2, nW1t, nW2t,
                                              (float*)d_out, N);
  int p3 = N * 3;
  pos_kernel<<<(p3 + 255) / 256, 256, 0, stream>>>(pos, agg_trans,
                                                   (float*)d_out + (size_t)N * HD, p3);
}
